// GATLayer_40106404610517
// MI455X (gfx1250) — matrix operand order repaired and checked by reading, not compiled, not run on hardware
//
#include <hip/hip_runtime.h>
#include <hip/hip_bf16.h>
#include <math.h>

// ---------------------------------------------------------------------------
// CDNA5 (gfx1250) wave32 WMMA types
// ---------------------------------------------------------------------------
typedef __attribute__((ext_vector_type(16))) __bf16 v16bf;
typedef __attribute__((ext_vector_type(8)))  float  v8f;

#define HID 128

// ---------------------------------------------------------------------------
// gfx1250 async global->LDS copy (ASYNCcnt-tracked), via inline asm.
// LDS operand is the workgroup-relative byte offset (low 32 bits of the flat
// shared-memory address).
// ---------------------------------------------------------------------------
__device__ __forceinline__ unsigned lds_off_u32(const void* p) {
  return (unsigned)(uintptr_t)p;
}

__device__ __forceinline__ void async_copy_b128(unsigned lds_byte, const void* gaddr) {
  asm volatile("global_load_async_to_lds_b128 %0, %1, off"
               :: "v"(lds_byte), "v"((unsigned long long)(uintptr_t)gaddr)
               : "memory");
}

__device__ __forceinline__ void wait_async0() {
  asm volatile("s_wait_asynccnt 0" ::: "memory");
}

// A fragment: 16x32 bf16, per CDNA5 layout.
// lanes 0-15 : M=lane,    K = {0..7} U {16..23}
// lanes 16-31: M=lane-16, K = {8..15} U {24..31}
__device__ __forceinline__ v16bf a_frag_from_lds(const float* Xs, int lane, int ldk) {
  const int row = lane & 15;
  const int ks  = (lane < 16) ? 0 : 8;
  const float* p = Xs + row * ldk + ks;
  v16bf a;
#pragma unroll
  for (int i = 0; i < 8; ++i) a[i] = (__bf16)p[i];
#pragma unroll
  for (int i = 0; i < 8; ++i) a[8 + i] = (__bf16)p[16 + i];
  return a;
}

// B fragment: 32x16 bf16 slice of a [32][128] fp32 LDS weight chunk.
// lanes 0-15 : N=col0+lane,    K = {0..7} U {16..23}
// lanes 16-31: N=col0+lane-16, K = {8..15} U {24..31}
__device__ __forceinline__ v16bf b_frag_from_lds(const float* Ws, int lane, int col0) {
  const int col = col0 + (lane & 15);
  const int ks  = (lane < 16) ? 0 : 8;
  const float* p = Ws + ks * HID + col;
  v16bf b;
#pragma unroll
  for (int i = 0; i < 16; ++i) b[i] = (__bf16)p[(i < 8 ? i : i + 8) * HID];
  return b;
}

__device__ __forceinline__ v8f wmma_bf16(v16bf a, v16bf b, v8f c) {
  return __builtin_amdgcn_wmma_f32_16x16x32_bf16(false, a, false, b, (short)0, c,
                                                 false, false);
}

// Block-level GEMM: Os[16][ldo](LDS) = act( Xs[16][K](LDS) @ Wg[K][128](global) + bias )
// 128 threads = 4 waves; wave w computes column tiles 2w, 2w+1.
// Weight chunks staged with async global->LDS b128 copies.
__device__ __forceinline__ void gemm16xK_128(const float* Xs, int K, int ldk,
                                             const float* __restrict__ Wg,
                                             const float* __restrict__ bias,
                                             float* Wstage /*LDS 32*128*/,
                                             float* Os, int ldo, bool relu, int tid) {
  const int wave = tid >> 5, lane = tid & 31;
  v8f acc0 = {}, acc1 = {};
  for (int k0 = 0; k0 < K; k0 += 32) {
#pragma unroll
    for (int j = 0; j < 8; ++j) {             // 32x128 fp32 chunk = 1024 float4
      int e = j * 128 + tid;
      async_copy_b128(lds_off_u32(&Wstage[e * 4]), Wg + (long)k0 * HID + e * 4);
    }
    wait_async0();
    __syncthreads();
    v16bf a  = a_frag_from_lds(Xs + k0, lane, ldk);
    v16bf b0 = b_frag_from_lds(Wstage, lane, wave * 32);
    v16bf b1 = b_frag_from_lds(Wstage, lane, wave * 32 + 16);
    acc0 = wmma_bf16(a, b0, acc0);
    acc1 = wmma_bf16(a, b1, acc1);
    __syncthreads();
  }
  const int rb = (lane >> 4) * 8;
  const int c0 = wave * 32 + (lane & 15);
  const int c1 = c0 + 16;
  const float bb0 = bias ? bias[c0] : 0.0f;
  const float bb1 = bias ? bias[c1] : 0.0f;
#pragma unroll
  for (int r = 0; r < 8; ++r) {
    float v0 = acc0[r] + bb0, v1 = acc1[r] + bb1;
    if (relu) { v0 = fmaxf(v0, 0.0f); v1 = fmaxf(v1, 0.0f); }
    Os[(rb + r) * ldo + c0] = v0;
    Os[(rb + r) * ldo + c1] = v1;
  }
  __syncthreads();
}

// ---------------------------------------------------------------------------
// Kernel 1: modal fusion (used for both img and txt paths)
//   feat = X @ Wproj ; g = sigmoid(MLP(concat[embed, feat]))
//   out_v = g*feat + (1-g)*embed
// ---------------------------------------------------------------------------
__global__ __launch_bounds__(128) void fuse_modal_kernel(
    const float* __restrict__ X, int D, const float* __restrict__ Wproj,
    const int* __restrict__ node_id, const float* __restrict__ emb,
    const float* __restrict__ W1, const float* __restrict__ b1,
    const float* __restrict__ W2, const float* __restrict__ b2,
    const float* __restrict__ W3, const float* __restrict__ b3,
    float* __restrict__ out_v, int N) {
  __shared__ float cat[16 * 256];     // [embed | feat]
  __shared__ float h1s[16 * 128];
  __shared__ float h2s[16 * 128];
  __shared__ float stage[32 * 128];
  __shared__ float atile[16 * 32];
  __shared__ float gsh[16];

  const int tid = threadIdx.x, wave = tid >> 5, lane = tid & 31;
  const int rbase = blockIdx.x * 16;

  // ---- feat = X[16 x D] @ Wproj[D x 128] via async-staged bf16 WMMA ----
  v8f acc0 = {}, acc1 = {};
  for (int k0 = 0; k0 < D; k0 += 32) {
    {                                          // A tile 16x32 = 128 float4
      int row = tid >> 3, q = tid & 7;
      int gr = rbase + row; if (gr >= N) gr = N - 1;
      async_copy_b128(lds_off_u32(&atile[row * 32 + q * 4]),
                      X + (long)gr * D + k0 + q * 4);
    }
#pragma unroll
    for (int j = 0; j < 8; ++j) {              // W chunk 32x128 = 1024 float4
      int e = j * 128 + tid;
      async_copy_b128(lds_off_u32(&stage[e * 4]), Wproj + (long)k0 * HID + e * 4);
    }
    wait_async0();
    __syncthreads();
    v16bf a  = a_frag_from_lds(atile, lane, 32);
    v16bf b0 = b_frag_from_lds(stage, lane, wave * 32);
    v16bf b1 = b_frag_from_lds(stage, lane, wave * 32 + 16);
    acc0 = wmma_bf16(a, b0, acc0);
    acc1 = wmma_bf16(a, b1, acc1);
    __syncthreads();
  }
  {
    const int rb = (lane >> 4) * 8;
    const int c0 = wave * 32 + (lane & 15), c1 = c0 + 16;
#pragma unroll
    for (int r = 0; r < 8; ++r) {
      cat[(rb + r) * 256 + 128 + c0] = acc0[r];
      cat[(rb + r) * 256 + 128 + c1] = acc1[r];
    }
  }
#pragma unroll
  for (int j = 0; j < 16; ++j) {              // embed gather into cat[:, :128]
    int e = j * 128 + tid;
    int row = e >> 7, col = e & 127;
    int gr = rbase + row; if (gr >= N) gr = N - 1;
    cat[row * 256 + col] = emb[(long)node_id[gr] * HID + col];
  }
  __syncthreads();

  // ---- gate MLP ----
  gemm16xK_128(cat, 256, 256, W1, b1, stage, h1s, 128, true, tid);
  gemm16xK_128(h1s, 128, 128, W2, b2, stage, h2s, 128, true, tid);
  if (tid < 16) {
    float s = b3[0];
#pragma unroll 4
    for (int k = 0; k < 128; ++k) s += h2s[tid * 128 + k] * W3[k];
    gsh[tid] = 1.0f / (1.0f + __expf(-s));
  }
  __syncthreads();

  // ---- out_v = g*feat + (1-g)*embed ----
#pragma unroll
  for (int j = 0; j < 16; ++j) {
    int e = j * 128 + tid;
    int row = e >> 7, col = e & 127;
    if (rbase + row < N) {
      float g = gsh[row];
      float f = cat[row * 256 + 128 + col];
      float m = cat[row * 256 + col];
      out_v[(long)(rbase + row) * HID + col] = g * f + (1.0f - g) * m;
    }
  }
}

// ---------------------------------------------------------------------------
// Kernel 2: combine gate + fc + attention scores
// ---------------------------------------------------------------------------
__global__ __launch_bounds__(128) void combine_kernel(
    const float* __restrict__ img_v, const float* __restrict__ txt_v,
    const float* __restrict__ comb_W, const float* __restrict__ comb_b,
    const float* __restrict__ fc_W, const float* __restrict__ attn_a,
    float* __restrict__ z, float* __restrict__ e_src, float* __restrict__ e_dst,
    int N) {
  __shared__ float cat[16 * 256];     // [img_v | txt_v]
  __shared__ float gate[16 * 128];
  __shared__ float hbuf[16 * 128];
  __shared__ float stage[32 * 128];
  __shared__ float ztile[16 * 128];

  const int tid = threadIdx.x;
  const int rbase = blockIdx.x * 16;

#pragma unroll
  for (int j = 0; j < 16; ++j) {
    int e = j * 128 + tid;
    int row = e >> 7, col = e & 127;
    int gr = rbase + row; if (gr >= N) gr = N - 1;
    cat[row * 256 + col]       = img_v[(long)gr * HID + col];
    cat[row * 256 + 128 + col] = txt_v[(long)gr * HID + col];
  }
  __syncthreads();

  gemm16xK_128(cat, 256, 256, comb_W, comb_b, stage, gate, 128, false, tid);

#pragma unroll
  for (int j = 0; j < 16; ++j) {
    int e = j * 128 + tid;
    int row = e >> 7, col = e & 127;
    float g = gate[row * 128 + col];
    hbuf[row * 128 + col] =
        g * cat[row * 256 + col] + (1.0f - g) * cat[row * 256 + 128 + col];
  }
  __syncthreads();

  gemm16xK_128(hbuf, 128, 128, fc_W, nullptr, stage, ztile, 128, false, tid);

#pragma unroll
  for (int j = 0; j < 16; ++j) {
    int e = j * 128 + tid;
    int row = e >> 7, col = e & 127;
    if (rbase + row < N) z[(long)(rbase + row) * HID + col] = ztile[row * 128 + col];
  }
  if (tid < 32) {
    int row = tid & 15;
    if (rbase + row < N) {
      const float* av = attn_a + ((tid < 16) ? 0 : HID);
      float s = 0.0f;
#pragma unroll 4
      for (int k = 0; k < 128; ++k) s += ztile[row * 128 + k] * av[k];
      if (tid < 16) e_src[rbase + row] = s; else e_dst[rbase + row] = s;
    }
  }
}

// ---------------------------------------------------------------------------
// Edge / segment-softmax kernels
// ---------------------------------------------------------------------------
__device__ __forceinline__ unsigned mval(float f) {
  unsigned u = __float_as_uint(f);
  return (u & 0x80000000u) ? ~u : (u | 0x80000000u);
}
__device__ __forceinline__ float unmval(unsigned m) {
  unsigned u = (m & 0x80000000u) ? (m & 0x7FFFFFFFu) : ~m;
  return __uint_as_float(u);
}

__global__ void init_kernel(float* out, unsigned* emax_u, float* denom, int N) {
  int i = blockIdx.x * blockDim.x + threadIdx.x;
  if (i < N * HID) out[i] = 0.0f;
  if (i < N) { emax_u[i] = 0u; denom[i] = 0.0f; }
}

__global__ void edge_max_kernel(const int* __restrict__ src, const int* __restrict__ dst,
                                const float* __restrict__ e_src, const float* __restrict__ e_dst,
                                float* __restrict__ e_edge, unsigned* __restrict__ emax_u, int E) {
  int i = blockIdx.x * blockDim.x + threadIdx.x;
  if (i >= E) return;
  float e = e_src[src[i]] + e_dst[dst[i]];
  e = (e > 0.0f) ? e : 0.01f * e;          // leaky_relu(0.01)
  e_edge[i] = e;
  atomicMax(&emax_u[dst[i]], mval(e));
}

__global__ void node_fix_kernel(const unsigned* __restrict__ emax_u,
                                float* __restrict__ emax_f, int N) {
  int i = blockIdx.x * blockDim.x + threadIdx.x;
  if (i >= N) return;
  float f = unmval(emax_u[i]);
  if (!isfinite(f)) f = 0.0f;              // segments with no edges
  emax_f[i] = f;
}

__global__ void edge_exp_kernel(const int* __restrict__ dst, const float* __restrict__ e_edge,
                                const float* __restrict__ emax_f, float* __restrict__ ex_edge,
                                float* __restrict__ denom, int E) {
  int i = blockIdx.x * blockDim.x + threadIdx.x;
  if (i >= E) return;
  int d = dst[i];
  float ex = __expf(e_edge[i] - emax_f[d]);
  ex_edge[i] = ex;
  unsafeAtomicAdd(&denom[d], ex);
}

// One wave per edge; lane handles 4 columns (float4 gather + 4 atomic fadds).
__global__ __launch_bounds__(256) void edge_scatter_kernel(
    const int* __restrict__ src, const int* __restrict__ dst,
    const float* __restrict__ ex_edge, const float* __restrict__ denom,
    const float* __restrict__ z, float* __restrict__ out, int E) {
  int w = blockIdx.x * 8 + (threadIdx.x >> 5);
  if (w >= E) return;
  int lane = threadIdx.x & 31;
  int s = src[w], d = dst[w];
  float alpha = ex_edge[w] / fmaxf(denom[d], 1e-20f);
  const float4 zz = *reinterpret_cast<const float4*>(z + (long)s * HID + lane * 4);
  float* op = out + (long)d * HID + lane * 4;
  unsafeAtomicAdd(op + 0, alpha * zz.x);
  unsafeAtomicAdd(op + 1, alpha * zz.y);
  unsafeAtomicAdd(op + 2, alpha * zz.z);
  unsafeAtomicAdd(op + 3, alpha * zz.w);
}

// ---------------------------------------------------------------------------
// Launch
// ---------------------------------------------------------------------------
extern "C" void kernel_launch(void* const* d_in, const int* in_sizes, int n_in,
                              void* d_out, int out_size, void* d_ws, size_t ws_size,
                              hipStream_t stream) {
  const int*   node_id = (const int*)d_in[0];
  const float* img_h   = (const float*)d_in[1];
  const float* txt_h   = (const float*)d_in[2];
  const int*   src     = (const int*)d_in[3];
  const int*   dst     = (const int*)d_in[4];
  const float* emb     = (const float*)d_in[5];
  const float* W_img   = (const float*)d_in[6];
  const float* img_W1  = (const float*)d_in[7];
  const float* img_b1  = (const float*)d_in[8];
  const float* img_W2  = (const float*)d_in[9];
  const float* img_b2  = (const float*)d_in[10];
  const float* img_W3  = (const float*)d_in[11];
  const float* img_b3  = (const float*)d_in[12];
  const float* W_txt   = (const float*)d_in[13];
  const float* txt_W1  = (const float*)d_in[14];
  const float* txt_b1  = (const float*)d_in[15];
  const float* txt_W2  = (const float*)d_in[16];
  const float* txt_b2  = (const float*)d_in[17];
  const float* txt_W3  = (const float*)d_in[18];
  const float* txt_b3  = (const float*)d_in[19];
  const float* comb_W  = (const float*)d_in[20];
  const float* comb_b  = (const float*)d_in[21];
  const float* fc_W    = (const float*)d_in[22];
  const float* attn_a  = (const float*)d_in[23];

  const int N = in_sizes[0];
  const int E = in_sizes[3];
  const int IMG_D = in_sizes[1] / N;
  const int TXT_D = in_sizes[2] / N;

  float* ws = (float*)d_ws;
  float*    z       = ws;                       // N*128
  float*    img_v   = z + (long)N * HID;        // N*128
  float*    txt_v   = img_v + (long)N * HID;    // N*128
  float*    e_src   = txt_v + (long)N * HID;    // N
  float*    e_dst   = e_src + N;                // N
  unsigned* emax_u  = (unsigned*)(e_dst + N);   // N
  float*    emax_f  = (float*)(emax_u + N);     // N
  float*    denom   = emax_f + N;               // N
  float*    e_edge  = denom + N;                // E
  float*    ex_edge = e_edge + E;               // E

  float* out = (float*)d_out;

  const int nb = (N + 15) / 16;

  init_kernel<<<((long)N * HID + 255) / 256, 256, 0, stream>>>(out, emax_u, denom, N);

  fuse_modal_kernel<<<nb, 128, 0, stream>>>(img_h, IMG_D, W_img, node_id, emb,
                                            img_W1, img_b1, img_W2, img_b2,
                                            img_W3, img_b3, img_v, N);
  fuse_modal_kernel<<<nb, 128, 0, stream>>>(txt_h, TXT_D, W_txt, node_id, emb,
                                            txt_W1, txt_b1, txt_W2, txt_b2,
                                            txt_W3, txt_b3, txt_v, N);
  combine_kernel<<<nb, 128, 0, stream>>>(img_v, txt_v, comb_W, comb_b, fc_W,
                                         attn_a, z, e_src, e_dst, N);

  edge_max_kernel<<<(E + 255) / 256, 256, 0, stream>>>(src, dst, e_src, e_dst,
                                                       e_edge, emax_u, E);
  node_fix_kernel<<<(N + 255) / 256, 256, 0, stream>>>(emax_u, emax_f, N);
  edge_exp_kernel<<<(E + 255) / 256, 256, 0, stream>>>(dst, e_edge, emax_f,
                                                       ex_edge, denom, E);
  edge_scatter_kernel<<<(E + 7) / 8, 256, 0, stream>>>(src, dst, ex_edge, denom,
                                                       z, out, E);
}
